// HumanNodeRNN_13950053777984
// MI455X (gfx1250) — matrix operand order repaired and checked by reading, not compiled, not run on hardware
//
#include <hip/hip_runtime.h>
#include <hip/hip_bf16.h>

typedef __attribute__((ext_vector_type(16))) _Float16 v16h;
typedef __attribute__((ext_vector_type(8)))  _Float16 v8h;
typedef __attribute__((ext_vector_type(8)))  float    v8f;
typedef __attribute__((ext_vector_type(4)))  float    v4f;

#define EMB   64
#define RNN   128
#define KHO   256   // 2*EDGE_RNN
#define G3    384   // 3*RNN
#define OUTW  5
#define BLK   512
#define NWAVE 16

__device__ __forceinline__ void sched_fence() {
  __builtin_amdgcn_sched_barrier(0);  // no instruction motion across this point
}

__device__ __forceinline__ v8f wmma16(v16h a, v16h b, v8f c) {
  return __builtin_amdgcn_wmma_f32_16x16x32_f16(false, a, false, b, (short)0, c,
                                                false, false);
}

// A-fragment (16x32 f16) from a global f32 row; kbase = kb*32 + khalf*8.
// elems 0..7  <- K = kbase..kbase+7 ; elems 8..15 <- K = kbase+16..kbase+23
__device__ __forceinline__ v16h load_afrag_global(const float* row, int kbase) {
  v4f a0 = *(const v4f*)(row + kbase);
  v4f a1 = *(const v4f*)(row + kbase + 4);
  v4f a2 = *(const v4f*)(row + kbase + 16);
  v4f a3 = *(const v4f*)(row + kbase + 20);
  v16h r;
#pragma unroll
  for (int i = 0; i < 4; ++i) {
    r[i]      = (_Float16)a0[i];
    r[4 + i]  = (_Float16)a1[i];
    r[8 + i]  = (_Float16)a2[i];
    r[12 + i] = (_Float16)a3[i];
  }
  return r;
}

// A-fragment from an f16 row in LDS (same element mapping as above)
__device__ __forceinline__ v16h load_afrag_lds(const _Float16* row, int kbase) {
  v8h lo = *(const v8h*)(row + kbase);
  v8h hi = *(const v8h*)(row + kbase + 16);
  v16h r;
#pragma unroll
  for (int i = 0; i < 8; ++i) { r[i] = lo[i]; r[8 + i] = hi[i]; }
  return r;
}

// B-fragment (32x16 f16): column n = lane&15; elems 0..7 <- K = kb*32+khalf*8..+7,
// elems 8..15 <- K = kb*32+16+khalf*8..+7 (same mapping as the A fragments).
// Weights stored row-major [Nout][K] as f16 in LDS, so this is two 16B reads.
__device__ __forceinline__ v16h load_bfrag(const _Float16* w, int nrow, int K,
                                           int kb, int khalf) {
  const _Float16* p = w + nrow * K + kb * 32 + khalf * 8;
  v8h lo = *(const v8h*)(p);
  v8h hi = *(const v8h*)(p + 16);
  v16h r;
#pragma unroll
  for (int i = 0; i < 8; ++i) { r[i] = lo[i]; r[8 + i] = hi[i]; }
  return r;
}

__global__ __launch_bounds__(BLK, 1) void gru_fused_kernel(
    const float* __restrict__ pos, const float* __restrict__ h_other,
    const float* __restrict__ h, const float* __restrict__ enc_W,
    const float* __restrict__ enc_b, const float* __restrict__ henc_W,
    const float* __restrict__ henc_b, const float* __restrict__ W_ih,
    const float* __restrict__ b_ih, const float* __restrict__ W_hh,
    const float* __restrict__ b_hh, const float* __restrict__ dec_W,
    const float* __restrict__ dec_b, float* __restrict__ out5,
    float* __restrict__ outh, int nTiles) {
  // ---- LDS: f16 weights + per-wave x/h_new staging (~296 KB of 320 KB) ----
  __shared__ _Float16 lds_henc[EMB * KHO];     // 32 KB
  __shared__ _Float16 lds_wih[G3 * RNN];       // 96 KB
  __shared__ _Float16 lds_whh[G3 * RNN];       // 96 KB
  __shared__ _Float16 lds_dec[16 * RNN];       //  4 KB (OUT padded to 16)
  __shared__ _Float16 lds_x[NWAVE * 16 * RNN]; // 64 KB
  __shared__ float lds_brz[2 * RNN];  // b_ih + b_hh for r,z gates (combined)
  __shared__ float lds_bxn[RNN], lds_bhn[RNN];
  __shared__ float lds_hencb[EMB], lds_encb[EMB], lds_encw[2 * EMB], lds_decb[16];

  const int tid = threadIdx.x;
  // stage + convert weights once per workgroup
  for (int i = tid; i < EMB * KHO; i += BLK) lds_henc[i] = (_Float16)henc_W[i];
  for (int i = tid; i < G3 * RNN; i += BLK) {
    lds_wih[i] = (_Float16)W_ih[i];
    lds_whh[i] = (_Float16)W_hh[i];
  }
  for (int i = tid; i < 16 * RNN; i += BLK) {
    int r = i >> 7, c = i & (RNN - 1);
    lds_dec[i] = (r < OUTW) ? (_Float16)dec_W[r * RNN + c] : (_Float16)0.f;
  }
  for (int i = tid; i < 2 * RNN; i += BLK) lds_brz[i] = b_ih[i] + b_hh[i];
  for (int i = tid; i < RNN; i += BLK) {
    lds_bxn[i] = b_ih[2 * RNN + i];
    lds_bhn[i] = b_hh[2 * RNN + i];
  }
  for (int i = tid; i < EMB; i += BLK) { lds_hencb[i] = henc_b[i]; lds_encb[i] = enc_b[i]; }
  for (int i = tid; i < 2 * EMB; i += BLK) lds_encw[i] = enc_W[i];
  if (tid < 16) lds_decb[tid] = (tid < OUTW) ? dec_b[tid] : 0.f;
  __syncthreads();

  const int lane  = tid & 31;
  const int wave  = tid >> 5;
  const int col   = lane & 15;  // N-column within 16-wide tile / A-row index
  const int khalf = lane >> 4;  // K-half selector
  _Float16* xbuf = &lds_x[wave * 16 * RNN];  // per-wave [16][128] f16

  for (int tile = blockIdx.x * NWAVE + wave; tile < nTiles;
       tile += gridDim.x * NWAVE) {
    const int r0 = tile << 4;

    // ---- enc_in = relu(pos @ enc_W.T + enc_b) -> xbuf[:, 0:64] (VALU, K=2)
    {
      const int grow = r0 + col;
      const float p0 = pos[grow * 2 + 0];
      const float p1 = pos[grow * 2 + 1];
      const int cbase = khalf * 32;
#pragma unroll
      for (int i = 0; i < 32; ++i) {
        const int c = cbase + i;
        float v = fmaf(p0, lds_encw[2 * c], fmaf(p1, lds_encw[2 * c + 1], lds_encb[c]));
        xbuf[col * RNN + c] = (_Float16)fmaxf(v, 0.f);
      }
    }
    sched_fence();

    // ---- enc_hid = relu(h_other @ henc_W.T + henc_b) -> xbuf[:, 64:128]
    {
      const float* arow = h_other + (size_t)(r0 + col) * KHO;
      v16h af[8];
#pragma unroll
      for (int kb = 0; kb < 8; ++kb)
        af[kb] = load_afrag_global(arow, kb * 32 + khalf * 8);
#pragma unroll
      for (int nt = 0; nt < 4; ++nt) {
        const float bias = lds_hencb[nt * 16 + col];
        v8f acc;
#pragma unroll
        for (int j = 0; j < 8; ++j) acc[j] = bias;
#pragma unroll
        for (int kb = 0; kb < 8; ++kb)
          acc = wmma16(af[kb], load_bfrag(lds_henc, nt * 16 + col, KHO, kb, khalf), acc);
#pragma unroll
        for (int j = 0; j < 8; ++j)  // D layout: row = j + 8*khalf, col = lane&15
          xbuf[(j + 8 * khalf) * RNN + EMB + nt * 16 + col] =
              (_Float16)fmaxf(acc[j], 0.f);
      }
    }
    sched_fence();

    // ---- resident A fragments: x (from LDS) and h (from global)
    const float* hrow = h + (size_t)(r0 + col) * RNN;
    v16h xf[4], hf[4];
#pragma unroll
    for (int kb = 0; kb < 4; ++kb) {
      xf[kb] = load_afrag_lds(xbuf + col * RNN, kb * 32 + khalf * 8);
      hf[kb] = load_afrag_global(hrow, kb * 32 + khalf * 8);
    }
    sched_fence();

    // ---- GRU gates + fused epilogue, 8 column tiles of 16.
    // r and z gates fold both GEMMs (x@W and h@U) into ONE accumulator each
    // (bias-initialized with b_ih+b_hh); only n keeps xn/hn separate.
#pragma unroll 1
    for (int t = 0; t < 8; ++t) {
      sched_fence();
      const int n0 = t * 16 + col;
      v8f ar, az, axn, ahn;
      {
        const float b0 = lds_brz[n0], b1 = lds_brz[RNN + n0];
        const float b2 = lds_bxn[n0], b3 = lds_bhn[n0];
#pragma unroll
        for (int j = 0; j < 8; ++j) {
          ar[j] = b0; az[j] = b1; axn[j] = b2; ahn[j] = b3;
        }
      }
#pragma unroll
      for (int kb = 0; kb < 4; ++kb) {
        ar  = wmma16(xf[kb], load_bfrag(lds_wih, n0,           RNN, kb, khalf), ar);
        ar  = wmma16(hf[kb], load_bfrag(lds_whh, n0,           RNN, kb, khalf), ar);
        az  = wmma16(xf[kb], load_bfrag(lds_wih, RNN + n0,     RNN, kb, khalf), az);
        az  = wmma16(hf[kb], load_bfrag(lds_whh, RNN + n0,     RNN, kb, khalf), az);
        axn = wmma16(xf[kb], load_bfrag(lds_wih, 2 * RNN + n0, RNN, kb, khalf), axn);
        ahn = wmma16(hf[kb], load_bfrag(lds_whh, 2 * RNN + n0, RNN, kb, khalf), ahn);
      }
#pragma unroll
      for (int j = 0; j < 8; ++j) {
        const int row  = j + 8 * khalf;
        const int grow = r0 + row;
        const float r = 1.f / (1.f + __expf(-ar[j]));
        const float z = 1.f / (1.f + __expf(-az[j]));
        const float pre = axn[j] + r * ahn[j];
        const float e2 = __expf(2.f * pre);
        const float nv = (e2 - 1.f) / (e2 + 1.f);
        const float hv = h[(size_t)grow * RNN + t * 16 + col];
        const float hnew = (1.f - z) * nv + z * hv;
        outh[(size_t)grow * RNN + t * 16 + col] = hnew;
        xbuf[row * RNN + t * 16 + col] = (_Float16)hnew;  // stage for decoder
      }
    }
    sched_fence();

    // ---- decoder: out = h_new @ dec_W.T + dec_b (OUT padded 5 -> 16)
    {
      const float bias = lds_decb[col];
      v8f acc;
#pragma unroll
      for (int j = 0; j < 8; ++j) acc[j] = bias;
#pragma unroll
      for (int kb = 0; kb < 4; ++kb) {
        v16h nf = load_afrag_lds(xbuf + col * RNN, kb * 32 + khalf * 8);
        acc = wmma16(nf, load_bfrag(lds_dec, col, RNN, kb, khalf), acc);
      }
      if (col < OUTW) {
#pragma unroll
        for (int j = 0; j < 8; ++j) {
          const int grow = r0 + j + 8 * khalf;
          out5[(size_t)grow * OUTW + col] = acc[j];
        }
      }
    }
    sched_fence();
  }
}

extern "C" void kernel_launch(void* const* d_in, const int* in_sizes, int n_in,
                              void* d_out, int out_size, void* d_ws, size_t ws_size,
                              hipStream_t stream) {
  const float* pos     = (const float*)d_in[0];
  const float* h_other = (const float*)d_in[1];
  const float* h       = (const float*)d_in[2];
  const float* enc_W   = (const float*)d_in[3];
  const float* enc_b   = (const float*)d_in[4];
  const float* henc_W  = (const float*)d_in[5];
  const float* henc_b  = (const float*)d_in[6];
  const float* W_ih    = (const float*)d_in[7];
  const float* b_ih    = (const float*)d_in[8];
  const float* W_hh    = (const float*)d_in[9];
  const float* b_hh    = (const float*)d_in[10];
  const float* dec_W   = (const float*)d_in[11];
  const float* dec_b   = (const float*)d_in[12];

  const int N = in_sizes[0] / 2;      // pos is [N, 2]
  float* out5 = (float*)d_out;        // [N, 5]
  float* outh = out5 + (size_t)N * OUTW;  // [N, 128]

  const int nTiles = N / 16;          // N = 500000 is divisible by 16
  int blocks = (nTiles + NWAVE - 1) / NWAVE;
  if (blocks > 512) blocks = 512;

  gru_fused_kernel<<<blocks, BLK, 0, stream>>>(
      pos, h_other, h, enc_W, enc_b, henc_W, henc_b, W_ih, b_ih, W_hh, b_hh,
      dec_W, dec_b, out5, outh, nTiles);
}
